// QueryConditionedRouter_12824772346167
// MI455X (gfx1250) — compile-verified, matrix operand order checked by reading, not run on hardware
//
#include <hip/hip_runtime.h>
#include <hip/hip_bf16.h>

#define HIDDEN      1024
#define QUERY       1024
#define NUM_EXPERTS 64
#define M_TOTAL     73728     // 128 * 576 rows
#define ROWS_PER_B  9216      // 16 * 576 (rows sharing one query b)
#define KSTEPS      (HIDDEN / 32)   // 32

// d_out float layout (return order: topk_scores, topk_idx, gate_scores, gate_logits)
#define OFF_TOPK    0
#define OFF_IDX     (M_TOTAL * 2)                      // 147456
#define OFF_SCORES  (M_TOTAL * 4)                      // 294912
#define OFF_LOGITS  (OFF_SCORES + M_TOTAL * 64)        // 5013504

typedef __attribute__((ext_vector_type(16))) __bf16 v16bf;
typedef __attribute__((ext_vector_type(8)))  float  v8f;

// ---------------------------------------------------------------------------
// Kernel 0: qry_logits[b][e] = query_emb[b,:] . W_gate[e, HIDDEN:]
// ---------------------------------------------------------------------------
__global__ void qry_logits_kernel(const float* __restrict__ q,
                                  const float* __restrict__ Wg,
                                  float* __restrict__ qlog) {
    const int e = threadIdx.x;          // 64 experts
    const int b = blockIdx.x;           // 8 queries
    const float* __restrict__ qr = q  + (size_t)b * QUERY;
    const float* __restrict__ wr = Wg + (size_t)e * (HIDDEN + QUERY) + HIDDEN;
    float s = 0.0f;
    #pragma unroll 4
    for (int k = 0; k < QUERY; k += 4) {
        float4 a = *(const float4*)(qr + k);
        float4 w = *(const float4*)(wr + k);
        s = fmaf(a.x, w.x, s); s = fmaf(a.y, w.y, s);
        s = fmaf(a.z, w.z, s); s = fmaf(a.w, w.w, s);
    }
    qlog[b * NUM_EXPERTS + e] = s;
}

// ---------------------------------------------------------------------------
// Kernel 1: pre-split W_vis into bf16 hi/lo, stored in B-fragment order:
//   wfrag[((kstep*4 + nt)*32 + lane)*32 + {0..15:hi, 16..31:lo}]
// 128 blocks (kstep*4+nt) x 32 lanes. Total 256 KB, stays hot in L2.
// ---------------------------------------------------------------------------
__global__ void wsplit_kernel(const float* __restrict__ Wg,
                              __bf16* __restrict__ wfrag) {
    const int lane  = threadIdx.x;          // 32
    const int kstep = blockIdx.x >> 2;
    const int nt    = blockIdx.x & 3;
    const int l16   = lane & 15;
    const int sel   = lane >> 4;            // 0: K {0-7,16-23}; 1: K {8-15,24-31}
    const int e     = nt * 16 + l16;

    const float* __restrict__ src =
        Wg + (size_t)e * (HIDDEN + QUERY) + kstep * 32 + sel * 8;
    __bf16* __restrict__ dst = wfrag + ((size_t)blockIdx.x * 32 + lane) * 32;

    float v[16];
    #pragma unroll
    for (int i = 0; i < 8; ++i)  v[i]     = src[i];        // K group 1
    #pragma unroll
    for (int i = 0; i < 8; ++i)  v[8 + i] = src[16 + i];   // K group 2
    #pragma unroll
    for (int i = 0; i < 16; ++i) {
        __bf16 h = (__bf16)v[i];
        dst[i]      = h;                                   // hi
        dst[16 + i] = (__bf16)(v[i] - (float)h);           // lo
    }
}

// split a float4 into bf16 hi/lo halves at fragment positions [base, base+4)
__device__ __forceinline__ void split4(const float4 x, v16bf& hi, v16bf& lo, int base) {
    float v[4] = {x.x, x.y, x.z, x.w};
    #pragma unroll
    for (int i = 0; i < 4; ++i) {
        __bf16 h = (__bf16)v[i];
        hi[base + i] = h;
        lo[base + i] = (__bf16)(v[i] - (float)h);
    }
}

#define WMMA_BF16(A, B, C) \
    __builtin_amdgcn_wmma_f32_16x16x32_bf16(false, (A), false, (B), (short)0, (C), false, false)

// ---------------------------------------------------------------------------
// Kernel 2: main router. 288 blocks x 256 threads (8 waves).
// Wave w computes rows [blk*256 + w*32, +32) x 64 experts via bf16x3 WMMA:
// two 16-row A fragments share one set of B fragments per K-step.
// ---------------------------------------------------------------------------
__global__ void __launch_bounds__(256)
router_kernel(const float* __restrict__ vis,     // (73728, 1024) fp32
              const __bf16* __restrict__ wfrag,  // pre-split B fragments
              const float* __restrict__ qlog,    // (8, 64)
              float* __restrict__ out) {
    __shared__ float s_tile[8 * 32 * 64];   // per-wave 32x64 logits (64 KB)
    __shared__ float s_max[8 * 32];
    __shared__ float s_rcp[8 * 32];

    const int lane  = threadIdx.x & 31;
    const int wave  = threadIdx.x >> 5;
    const int tileM = blockIdx.x * 256 + wave * 32;
    const int l16   = lane & 15;
    const int sel   = lane >> 4;            // 0: K {0-7,16-23}; 1: K {8-15,24-31}

    const float* __restrict__ Ap0 = vis + (size_t)(tileM      + l16) * HIDDEN + sel * 8;
    const float* __restrict__ Ap1 = vis + (size_t)(tileM + 16 + l16) * HIDDEN + sel * 8;
    const __bf16* __restrict__ Bp = wfrag + (size_t)lane * 32;

    v8f acc[8];   // [0..3]: rows 0-15 x nt;  [4..7]: rows 16-31 x nt
    #pragma unroll
    for (int i = 0; i < 8; ++i) acc[i] = (v8f){0,0,0,0,0,0,0,0};

    for (int ks = 0; ks < KSTEPS; ++ks) {
        // ---- two A fragments (fp32 -> bf16 hi/lo split on the fly) ----
        const float* a0 = Ap0 + ks * 32;
        const float* a1 = Ap1 + ks * 32;
        __builtin_prefetch(a0 + 32, 0, 0);   // global_prefetch next K chunk
        __builtin_prefetch(a1 + 32, 0, 0);
        v16bf a0h, a0l, a1h, a1l;
        split4(*(const float4*)(a0 +  0), a0h, a0l,  0);
        split4(*(const float4*)(a0 +  4), a0h, a0l,  4);
        split4(*(const float4*)(a0 + 16), a0h, a0l,  8);
        split4(*(const float4*)(a0 + 20), a0h, a0l, 12);
        split4(*(const float4*)(a1 +  0), a1h, a1l,  0);
        split4(*(const float4*)(a1 +  4), a1h, a1l,  4);
        split4(*(const float4*)(a1 + 16), a1h, a1l,  8);
        split4(*(const float4*)(a1 + 20), a1h, a1l, 12);

        // ---- B fragments: straight loads of pre-split bf16 (no VALU) ----
        const __bf16* bk = Bp + (size_t)ks * (4 * 32 * 32);
        const v16bf b0h = *(const v16bf*)(bk + 0 * 32 * 32);
        const v16bf b0l = *(const v16bf*)(bk + 0 * 32 * 32 + 16);
        const v16bf b1h = *(const v16bf*)(bk + 1 * 32 * 32);
        const v16bf b1l = *(const v16bf*)(bk + 1 * 32 * 32 + 16);
        const v16bf b2h = *(const v16bf*)(bk + 2 * 32 * 32);
        const v16bf b2l = *(const v16bf*)(bk + 2 * 32 * 32 + 16);
        const v16bf b3h = *(const v16bf*)(bk + 3 * 32 * 32);
        const v16bf b3l = *(const v16bf*)(bk + 3 * 32 * 32 + 16);

        // bf16x3 error-compensated accumulation (~fp32 accuracy)
        acc[0] = WMMA_BF16(a0h, b0h, acc[0]);
        acc[1] = WMMA_BF16(a0h, b1h, acc[1]);
        acc[2] = WMMA_BF16(a0h, b2h, acc[2]);
        acc[3] = WMMA_BF16(a0h, b3h, acc[3]);
        acc[4] = WMMA_BF16(a1h, b0h, acc[4]);
        acc[5] = WMMA_BF16(a1h, b1h, acc[5]);
        acc[6] = WMMA_BF16(a1h, b2h, acc[6]);
        acc[7] = WMMA_BF16(a1h, b3h, acc[7]);
        acc[0] = WMMA_BF16(a0h, b0l, acc[0]);
        acc[1] = WMMA_BF16(a0h, b1l, acc[1]);
        acc[2] = WMMA_BF16(a0h, b2l, acc[2]);
        acc[3] = WMMA_BF16(a0h, b3l, acc[3]);
        acc[4] = WMMA_BF16(a1h, b0l, acc[4]);
        acc[5] = WMMA_BF16(a1h, b1l, acc[5]);
        acc[6] = WMMA_BF16(a1h, b2l, acc[6]);
        acc[7] = WMMA_BF16(a1h, b3l, acc[7]);
        acc[0] = WMMA_BF16(a0l, b0h, acc[0]);
        acc[1] = WMMA_BF16(a0l, b1h, acc[1]);
        acc[2] = WMMA_BF16(a0l, b2h, acc[2]);
        acc[3] = WMMA_BF16(a0l, b3h, acc[3]);
        acc[4] = WMMA_BF16(a1l, b0h, acc[4]);
        acc[5] = WMMA_BF16(a1l, b1h, acc[5]);
        acc[6] = WMMA_BF16(a1l, b2h, acc[6]);
        acc[7] = WMMA_BF16(a1l, b3h, acc[7]);
    }

    // ---- epilogue: add query bias, stage logits to LDS ----
    float* tile = s_tile + wave * (32 * 64);
    const int bq = tileM / ROWS_PER_B;      // uniform per tile (9216 = 288*32)
    #pragma unroll
    for (int nt = 0; nt < 4; ++nt) {
        const int e = nt * 16 + l16;
        const float qv = qlog[bq * NUM_EXPERTS + e];
        #pragma unroll
        for (int j = 0; j < 8; ++j) {       // C layout: VGPR j -> row j + 8*sel
            tile[(     j + sel * 8) * 64 + e] = acc[nt    ][j] + qv;
            tile[(16 + j + sel * 8) * 64 + e] = acc[nt + 4][j] + qv;
        }
    }
    __syncthreads();

    // ---- softmax stats + top-2: one row per lane, all 32 lanes active ----
    {
        const float* rp = tile + lane * 64;
        float mx = rp[0];
        #pragma unroll 8
        for (int e = 1; e < 64; ++e) mx = fmaxf(mx, rp[e]);
        float sum = 0.0f, v1 = -1.0f, v2 = -1.0f;
        int i1 = 0, i2 = 0;
        for (int e = 0; e < 64; ++e) {
            const float ex = __expf(rp[e] - mx);
            sum += ex;
            if (ex > v1)      { v2 = v1; i2 = i1; v1 = ex; i1 = e; }
            else if (ex > v2) { v2 = ex; i2 = e; }
        }
        const float rcp = 1.0f / sum;
        s_max[wave * 32 + lane] = mx;
        s_rcp[wave * 32 + lane] = rcp;
        const size_t m = (size_t)tileM + lane;
        out[OFF_TOPK + m * 2 + 0] = v1 * rcp;
        out[OFF_TOPK + m * 2 + 1] = v2 * rcp;
        out[OFF_IDX  + m * 2 + 0] = (float)i1;
        out[OFF_IDX  + m * 2 + 1] = (float)i2;
    }
    __syncthreads();

    // ---- coalesced streaming of gate_logits / gate_scores ----
    float* __restrict__ glog = out + OFF_LOGITS + (size_t)tileM * 64;
    float* __restrict__ gsc  = out + OFF_SCORES + (size_t)tileM * 64;
    #pragma unroll 8
    for (int i = 0; i < 64; ++i) {
        const int t   = i * 32 + lane;      // 32*64 = 2048 elements / wave
        const int row = t >> 6;
        const float x = tile[t];
        glog[t] = x;
        gsc[t]  = __expf(x - s_max[wave * 32 + row]) * s_rcp[wave * 32 + row];
    }
}

extern "C" void kernel_launch(void* const* d_in, const int* in_sizes, int n_in,
                              void* d_out, int out_size, void* d_ws, size_t ws_size,
                              hipStream_t stream) {
    const float* vis = (const float*)d_in[0];   // (128, 576, 1024)
    const float* qry = (const float*)d_in[1];   // (8, 1024)
    const float* Wg  = (const float*)d_in[2];   // (64, 2048)
    float* out = (float*)d_out;

    // workspace: [0, 2KB) qry_logits; [2KB, 2KB+256KB) pre-split W fragments
    float*  qlog  = (float*)d_ws;
    __bf16* wfrag = (__bf16*)((char*)d_ws + 2048);

    qry_logits_kernel<<<8, 64, 0, stream>>>(qry, Wg, qlog);
    wsplit_kernel<<<KSTEPS * 4, 32, 0, stream>>>(Wg, wfrag);
    router_kernel<<<M_TOTAL / 256, 256, 0, stream>>>(vis, wfrag, qlog, out);
}
